// UnifiedAttentionAggregator_56495999812125
// MI455X (gfx1250) — matrix operand order repaired and checked by reading, not compiled, not run on hardware
//
#include <hip/hip_runtime.h>
#include <hip/hip_bf16.h>

#define NINST 524288
#define NF    256
#define LDIM  128
#define NBAGS 4096

typedef __attribute__((ext_vector_type(16))) __bf16 v16bf;
typedef __attribute__((ext_vector_type(8)))  float  v8f;

// 17 column tiles (8 for V, 8 for U, 1 for W_ins padded), 8 K-chunks of 32.
#define NTILES 17
#define NCHUNK 8
#define WPACK_ELEMS (NTILES * NCHUNK * 32 * 16)   // 69632 bf16

// ---------------------------------------------------------------------------
// Kernel 1: repack weights into bf16 WMMA B-fragment layout in workspace.
// Fragment layout (ISA 7.12.2, 16-bit B 32x16): lane<16 -> col=lane, K=base+{0..7,16..23};
// lane>=16 -> col=lane-16, K=base+{8..15,24..31}  (e = element 0..15 within v16bf).
// ---------------------------------------------------------------------------
__global__ void pack_weights_kernel(const float* __restrict__ W_ins,
                                    const float* __restrict__ W_V,
                                    const float* __restrict__ W_U,
                                    __bf16* __restrict__ wsB) {
  int idx = blockIdx.x * blockDim.x + threadIdx.x;
  if (idx >= WPACK_ELEMS) return;
  int e    = idx & 15;
  int lane = (idx >> 4) & 31;
  int c    = (idx >> 9) & 7;     // K chunk
  int t    = idx >> 12;          // column tile
  int k    = c * 32 + ((e & 8) << 1) + ((lane >= 16) ? 8 : 0) + (e & 7);
  int col  = lane & 15;
  float v;
  if (t < 8)        v = W_V[k * LDIM + t * 16 + col];
  else if (t < 16)  v = W_U[k * LDIM + (t - 8) * 16 + col];
  else              v = (col == 0) ? W_ins[k] : 0.0f;
  wsB[idx] = (__bf16)v;
}

// ---------------------------------------------------------------------------
// Kernel 2: exclusive prefix sum of bag sizes (4096 = 256 threads x 16).
// ---------------------------------------------------------------------------
__global__ void scan_sizes_kernel(const long long* __restrict__ sizes,
                                  int* __restrict__ offs) {
  __shared__ int csum[256];
  const int t = threadIdx.x;
  const int base = t * 16;
  int local[16];
  int s = 0;
#pragma unroll
  for (int i = 0; i < 16; i++) { local[i] = s; s += (int)sizes[base + i]; }
  csum[t] = s;
  __syncthreads();
  for (int d = 1; d < 256; d <<= 1) {
    int v = (t >= d) ? csum[t - d] : 0;
    __syncthreads();
    csum[t] += v;
    __syncthreads();
  }
  int prev = (t == 0) ? 0 : csum[t - 1];
#pragma unroll
  for (int i = 0; i < 16; i++) offs[base + i] = prev + local[i];
}

// ---------------------------------------------------------------------------
// Kernel 3: fused streaming GEMM + gated attention scores + instance preds.
// One wave32 per 16 rows. 136 v_wmma_f32_16x16x32_bf16 per wave.
// ---------------------------------------------------------------------------
__global__ void __launch_bounds__(256)
mil_fused_gemm_kernel(const float* __restrict__ feats,
                      const __bf16* __restrict__ wsB,
                      const float* __restrict__ bV,
                      const float* __restrict__ bU,
                      const float* __restrict__ watt,
                      const float* __restrict__ batt,
                      const float* __restrict__ bins,
                      float* __restrict__ scores,
                      float* __restrict__ instp) {
  const int lane = threadIdx.x & 31;
  const int wv   = threadIdx.x >> 5;
  const int gw   = blockIdx.x * 8 + wv;
  const int m0   = gw << 4;                 // first of 16 rows for this wave
  const int half = lane >> 4;               // 0 or 1
  const int col  = lane & 15;
  const float* __restrict__ fr = feats + (size_t)(m0 + col) * NF;
  const v16bf* __restrict__ Bp = (const v16bf*)wsB;

  // --- Load A fragments: 16-bit A 16x32 layout (ISA 7.12.2):
  // lane<16: K = base+{0..7, 16..23}; lane>=16: K = base+{8..15, 24..31}.
  v16bf A[NCHUNK];
#pragma unroll
  for (int c = 0; c < NCHUNK; c++) {
    const int base = c * 32 + half * 8;
    const float4 f0 = *(const float4*)(fr + base);
    const float4 f1 = *(const float4*)(fr + base + 4);
    const float4 g0 = *(const float4*)(fr + base + 16);
    const float4 g1 = *(const float4*)(fr + base + 20);
    v16bf a;
    a[0]  = (__bf16)f0.x; a[1]  = (__bf16)f0.y; a[2]  = (__bf16)f0.z; a[3]  = (__bf16)f0.w;
    a[4]  = (__bf16)f1.x; a[5]  = (__bf16)f1.y; a[6]  = (__bf16)f1.z; a[7]  = (__bf16)f1.w;
    a[8]  = (__bf16)g0.x; a[9]  = (__bf16)g0.y; a[10] = (__bf16)g0.z; a[11] = (__bf16)g0.w;
    a[12] = (__bf16)g1.x; a[13] = (__bf16)g1.y; a[14] = (__bf16)g1.z; a[15] = (__bf16)g1.w;
    A[c] = a;
  }

  // partial attention-score accumulators (per lane: 8 rows x its 8 columns)
  float ps[8] = {0.f, 0.f, 0.f, 0.f, 0.f, 0.f, 0.f, 0.f};

  for (int j = 0; j < 8; j++) {               // 16-wide column blocks of L
    v8f accV = (v8f)0.0f;
    v8f accU = (v8f)0.0f;
#pragma unroll
    for (int c = 0; c < NCHUNK; c++) {
      v16bf b = Bp[(j * NCHUNK + c) * 32 + lane];
      accV = __builtin_amdgcn_wmma_f32_16x16x32_bf16(
          false, A[c], false, b, (short)0, accV, false, false);
    }
#pragma unroll
    for (int c = 0; c < NCHUNK; c++) {
      v16bf b = Bp[((j + 8) * NCHUNK + c) * 32 + lane];
      accU = __builtin_amdgcn_wmma_f32_16x16x32_bf16(
          false, A[c], false, b, (short)0, accU, false, false);
    }
    const int n   = j * 16 + col;
    const float bv = bV[n];
    const float bu = bU[n];
    const float wa = watt[n];
#pragma unroll
    for (int r = 0; r < 8; r++) {
      float g = tanhf(accV[r] + bv) *
                (1.0f / (1.0f + __expf(-(accU[r] + bu))));
      ps[r] += g * wa;
    }
  }

  // --- instance classifier tile (column 0 of tile 16 is W_ins) ---
  v8f accI = (v8f)0.0f;
#pragma unroll
  for (int c = 0; c < NCHUNK; c++) {
    v16bf b = Bp[(16 * NCHUNK + c) * 32 + lane];
    accI = __builtin_amdgcn_wmma_f32_16x16x32_bf16(
        false, A[c], false, b, (short)0, accI, false, false);
  }

  // --- reduce partial scores across the 16 lanes that share each row ---
  const float ba = batt[0];
#pragma unroll
  for (int r = 0; r < 8; r++) {
    float v = ps[r];
    v += __shfl_xor(v, 1, 32);
    v += __shfl_xor(v, 2, 32);
    v += __shfl_xor(v, 4, 32);
    v += __shfl_xor(v, 8, 32);
    ps[r] = v + ba;
  }

  // C layout: VGPR r -> row m0+r (lanes 0-15) / m0+8+r (lanes 16-31); N=col.
  if (col == 0) {
    const int rbase = m0 + half * 8;
    float4 s0 = {ps[0], ps[1], ps[2], ps[3]};
    float4 s1 = {ps[4], ps[5], ps[6], ps[7]};
    *(float4*)(scores + rbase)     = s0;
    *(float4*)(scores + rbase + 4) = s1;
    const float bi = bins[0];
    float4 i0 = {accI[0] + bi, accI[1] + bi, accI[2] + bi, accI[3] + bi};
    float4 i1 = {accI[4] + bi, accI[5] + bi, accI[6] + bi, accI[7] + bi};
    *(float4*)(instp + rbase)     = i0;
    *(float4*)(instp + rbase + 4) = i1;
  }
}

// ---------------------------------------------------------------------------
// Kernel 4: per-bag softmax + attention-weighted sum. One block per bag.
// ---------------------------------------------------------------------------
__global__ void __launch_bounds__(128)
bag_softmax_kernel(const float* __restrict__ scores,
                   const float* __restrict__ instp,
                   const long long* __restrict__ sizes,
                   const int* __restrict__ offs,
                   float* __restrict__ bag_pred) {
  __shared__ float red[128];
  const int b   = blockIdx.x;
  const int tid = threadIdx.x;
  const int off = offs[b];
  const int sz  = (int)sizes[b];

  float m = -3.402823466e38f;
  for (int i = tid; i < sz; i += 128) m = fmaxf(m, scores[off + i]);
  red[tid] = m;
  __syncthreads();
  for (int s = 64; s > 0; s >>= 1) {
    if (tid < s) red[tid] = fmaxf(red[tid], red[tid + s]);
    __syncthreads();
  }
  m = red[0];
  __syncthreads();

  float se = 0.f, sw = 0.f;
  for (int i = tid; i < sz; i += 128) {
    float e = __expf(scores[off + i] - m);
    se += e;
    sw += e * instp[off + i];
  }
  red[tid] = se;
  __syncthreads();
  for (int s = 64; s > 0; s >>= 1) {
    if (tid < s) red[tid] += red[tid + s];
    __syncthreads();
  }
  const float denom = red[0];
  __syncthreads();
  red[tid] = sw;
  __syncthreads();
  for (int s = 64; s > 0; s >>= 1) {
    if (tid < s) red[tid] += red[tid + s];
    __syncthreads();
  }
  if (tid == 0) bag_pred[b] = red[0] / denom;
}

// ---------------------------------------------------------------------------
extern "C" void kernel_launch(void* const* d_in, const int* in_sizes, int n_in,
                              void* d_out, int out_size, void* d_ws, size_t ws_size,
                              hipStream_t stream) {
  const float*      feats  = (const float*)d_in[0];
  const long long*  sizes  = (const long long*)d_in[1];
  const float*      W_ins  = (const float*)d_in[2];
  const float*      b_ins  = (const float*)d_in[3];
  const float*      W_V    = (const float*)d_in[4];
  const float*      b_V    = (const float*)d_in[5];
  const float*      W_U    = (const float*)d_in[6];
  const float*      b_U    = (const float*)d_in[7];
  const float*      w_att  = (const float*)d_in[8];
  const float*      b_att  = (const float*)d_in[9];

  float* bag_pred = (float*)d_out;            // [4096]
  float* instp    = (float*)d_out + NBAGS;    // [524288]

  char*   ws     = (char*)d_ws;
  __bf16* wsB    = (__bf16*)ws;                         // 139264 B
  int*    offs   = (int*)(ws + 139264);                 // 16384 B
  float*  scores = (float*)(ws + 139264 + 16384);       // 2 MB

  pack_weights_kernel<<<(WPACK_ELEMS + 255) / 256, 256, 0, stream>>>(
      W_ins, W_V, W_U, wsB);
  scan_sizes_kernel<<<1, 256, 0, stream>>>(sizes, offs);
  mil_fused_gemm_kernel<<<NINST / 128, 256, 0, stream>>>(
      feats, wsB, b_V, b_U, w_att, b_att, b_ins, scores, instp);
  bag_softmax_kernel<<<NBAGS, 128, 0, stream>>>(
      scores, instp, sizes, offs, bag_pred);
}
